// DiaSelfAttention_67688684585515
// MI455X (gfx1250) — compile-verified, matrix operand order checked by reading, not run on hardware
//
#include <hip/hip_runtime.h>
#include <hip/hip_bf16.h>

// ---------------------------------------------------------------------------
// GQA attention for B=2, S=2048, H=2048, NH=16, NKV=4, HD=128 on gfx1250.
// bf16 WMMA (v_wmma_f32_16x16x32_bf16); software-pipelined, double-buffered
// LDS GEMM tiles. attn output is the bandwidth-critical path (537 MB write
// mandated by the reference output). Workspace requirement: ~88 MB.
// ---------------------------------------------------------------------------

#define kB   2
#define kS   2048
#define kH   2048
#define kNH  16
#define kNKV 4
#define kHD  128
#define SCALE 0.08838834764831845f   // 1/sqrt(128)

typedef __bf16 bf16;
typedef __bf16 v16bf __attribute__((ext_vector_type(16)));
typedef __bf16 v8bf  __attribute__((ext_vector_type(8)));
typedef __bf16 v4bf  __attribute__((ext_vector_type(4)));
typedef float  v8f   __attribute__((ext_vector_type(8)));
typedef float  v4f   __attribute__((ext_vector_type(4)));

#define LDSS 40   // LDS row stride in bf16 elems (32 data + 8 pad -> 80B rows)

// ---------------------------------------------------------------------------
// WMMA wrapper: D = A(16x32 bf16) x B(32x16 bf16) + C(16x16 f32)
// ---------------------------------------------------------------------------
__device__ __forceinline__ v8f wmma_bf16(v16bf a, v16bf b, v8f c) {
  return __builtin_amdgcn_wmma_f32_16x16x32_bf16(
      /*neg_a=*/false, a, /*neg_b=*/false, b,
      /*c_mod=*/(short)0, c, /*reuse_a=*/false, /*reuse_b=*/false);
}

// Fragment load from LDS per ISA 7.12.2 16-bit A/B layout:
// lane L: row = row0 + (L&15); K-halves {0..7,16..23} (L<16) / {8..15,24..31}.
__device__ __forceinline__ v16bf frag_lds(const bf16* s, int row0, int lane) {
  int r = row0 + (lane & 15);
  int koff = (lane & 16) ? 8 : 0;
  const bf16* p = s + r * LDSS + koff;
  v8bf lo = *(const v8bf*)p;          // K = koff .. koff+7
  v8bf hi = *(const v8bf*)(p + 16);   // K = koff+16 .. koff+23
  return __builtin_shufflevector(lo, hi, 0,1,2,3,4,5,6,7,8,9,10,11,12,13,14,15);
}

// ---------------------------------------------------------------------------
// Core 128x128 tile GEMM: C[m,n] = sum_k A[m,k] * B[n,k]
// (both A and B row-major with K contiguous; B is the "N x K" operand).
// 8 waves; wave (wm,wn) owns a 64x32 sub-tile = 4x2 WMMA accumulators.
// Software pipeline: global loads for tile t+1 issue before the WMMA work on
// tile t; LDS is double-buffered so there is one barrier per K-step.
// ---------------------------------------------------------------------------
template<bool AF32>
__device__ __forceinline__ void gemm_tile(const void* __restrict__ A, long lda, int arow0,
                                          const bf16* __restrict__ Bm, long ldb, int brow0,
                                          int kdim,
                                          bf16* sA0, bf16* sB0, bf16* sA1, bf16* sB1,
                                          v8f acc[4][2]) {
  const int tid  = threadIdx.x;
  const int lane = tid & 31;
  const int wave = tid >> 5;
  const int wm = (wave & 1) << 6;          // 0 / 64
  const int wn = (wave >> 1) << 5;         // 0 / 32 / 64 / 96
  // Branchless staging: thread owns chunks (crow, ccol) and (crow+64, ccol),
  // each 8 bf16 (16B). 256 threads cover a full 128x32 tile per operand.
  const int crow = tid >> 2;
  const int ccol = (tid & 3) << 3;

  v8bf ra0, ra1;            // A chunks (bf16 path)
  v4f  fa0, fa1, fa2, fa3;  // A chunks (f32 path, converted at LDS-store time)
  v8bf rb0, rb1;            // B chunks

  auto load_g = [&](int k0) {
    if constexpr (AF32) {
      const float* Af = (const float*)A;
      const float* g0 = Af + (size_t)(arow0 + crow) * lda + k0 + ccol;
      const float* g1 = Af + (size_t)(arow0 + crow + 64) * lda + k0 + ccol;
      fa0 = *(const v4f*)g0; fa1 = *(const v4f*)(g0 + 4);
      fa2 = *(const v4f*)g1; fa3 = *(const v4f*)(g1 + 4);
    } else {
      const bf16* Ab = (const bf16*)A;
      ra0 = *(const v8bf*)(Ab + (size_t)(arow0 + crow) * lda + k0 + ccol);
      ra1 = *(const v8bf*)(Ab + (size_t)(arow0 + crow + 64) * lda + k0 + ccol);
    }
    rb0 = *(const v8bf*)(Bm + (size_t)(brow0 + crow) * ldb + k0 + ccol);
    rb1 = *(const v8bf*)(Bm + (size_t)(brow0 + crow + 64) * ldb + k0 + ccol);
  };

  auto prefetch_g = [&](int k0) {          // -> global_prefetch_b8
    if constexpr (AF32) {
      const float* Af = (const float*)A;
      __builtin_prefetch(Af + (size_t)(arow0 + crow) * lda + k0 + ccol, 0, 3);
      __builtin_prefetch(Af + (size_t)(arow0 + crow + 64) * lda + k0 + ccol, 0, 3);
    } else {
      const bf16* Ab = (const bf16*)A;
      __builtin_prefetch(Ab + (size_t)(arow0 + crow) * lda + k0 + ccol, 0, 3);
      __builtin_prefetch(Ab + (size_t)(arow0 + crow + 64) * lda + k0 + ccol, 0, 3);
    }
  };

  auto store_s = [&](bf16* sA, bf16* sB) {
    if constexpr (AF32) {
      v8bf v0, v1;
      v0[0] = (bf16)fa0[0]; v0[1] = (bf16)fa0[1]; v0[2] = (bf16)fa0[2]; v0[3] = (bf16)fa0[3];
      v0[4] = (bf16)fa1[0]; v0[5] = (bf16)fa1[1]; v0[6] = (bf16)fa1[2]; v0[7] = (bf16)fa1[3];
      v1[0] = (bf16)fa2[0]; v1[1] = (bf16)fa2[1]; v1[2] = (bf16)fa2[2]; v1[3] = (bf16)fa2[3];
      v1[4] = (bf16)fa3[0]; v1[5] = (bf16)fa3[1]; v1[6] = (bf16)fa3[2]; v1[7] = (bf16)fa3[3];
      *(v8bf*)(sA + crow * LDSS + ccol) = v0;
      *(v8bf*)(sA + (crow + 64) * LDSS + ccol) = v1;
    } else {
      *(v8bf*)(sA + crow * LDSS + ccol) = ra0;
      *(v8bf*)(sA + (crow + 64) * LDSS + ccol) = ra1;
    }
    *(v8bf*)(sB + crow * LDSS + ccol) = rb0;
    *(v8bf*)(sB + (crow + 64) * LDSS + ccol) = rb1;
  };

  const int nk = kdim >> 5;
  bf16 *cA = sA0, *cB = sB0, *nA = sA1, *nB = sB1;

  load_g(0);
  store_s(cA, cB);
  __syncthreads();

  for (int t = 0; t < nk; ++t) {
    if (t + 1 < nk) load_g((t + 1) << 5);      // issue next tile's loads early
    if (t + 2 < nk) prefetch_g((t + 2) << 5);  // deep prefetch of A stream
    v16bf b0 = frag_lds(cB, wn,      lane);
    v16bf b1 = frag_lds(cB, wn + 16, lane);
    #pragma unroll
    for (int i = 0; i < 4; ++i) {
      v16bf a = frag_lds(cA, wm + (i << 4), lane);
      acc[i][0] = wmma_bf16(a, b0, acc[i][0]);
      acc[i][1] = wmma_bf16(a, b1, acc[i][1]);
    }
    if (t + 1 < nk) store_s(nA, nB);           // waits on globals here only
    bf16* tp;
    tp = cA; cA = nA; nA = tp;
    tp = cB; cB = nB; nB = tp;
    __syncthreads();
  }
}

// C/D layout (ISA 7.12.2): acc[i][j] vgpr v -> row = wm+i*16+v+8*(lane>>4),
//                                              col = wn+j*16+(lane&15).
#define EPILOGUE_COORDS                                        \
  int tid  = threadIdx.x;                                      \
  int lane = tid & 31;                                         \
  int wave = tid >> 5;                                         \
  int r0 = ((wave & 1) << 6) + ((lane >> 4) << 3);             \
  int c0 = ((wave >> 1) << 5) + (lane & 15);

#define GEMM_SMEM                                              \
  __shared__ __align__(16) bf16 sA0[128 * LDSS];               \
  __shared__ __align__(16) bf16 sB0[128 * LDSS];               \
  __shared__ __align__(16) bf16 sA1[128 * LDSS];               \
  __shared__ __align__(16) bf16 sB1[128 * LDSS];

// ---------------------------------------------------------------------------
// Kernel: generic GEMM, bf16 output (QKV projections)
// ---------------------------------------------------------------------------
__global__ void __launch_bounds__(256)
gemm_to_bf16(const bf16* __restrict__ A, long lda,
             const bf16* __restrict__ Bm, long ldb,
             bf16* __restrict__ C, long ldc, int kdim) {
  GEMM_SMEM
  int row0 = blockIdx.y << 7, col0 = blockIdx.x << 7;
  v8f acc[4][2] = {};
  gemm_tile<false>(A, lda, row0, Bm, ldb, col0, kdim, sA0, sB0, sA1, sB1, acc);
  EPILOGUE_COORDS
  #pragma unroll
  for (int i = 0; i < 4; ++i)
    #pragma unroll
    for (int j = 0; j < 2; ++j)
      #pragma unroll
      for (int v = 0; v < 8; ++v)
        C[(size_t)(row0 + r0 + (i << 4) + v) * ldc + col0 + c0 + (j << 4)] =
            (bf16)acc[i][j][v];
}

// ---------------------------------------------------------------------------
// Kernel: generic GEMM, f32 output (output projection)
// ---------------------------------------------------------------------------
__global__ void __launch_bounds__(256)
gemm_to_f32(const bf16* __restrict__ A, long lda,
            const bf16* __restrict__ Bm, long ldb,
            float* __restrict__ C, long ldc, int kdim) {
  GEMM_SMEM
  int row0 = blockIdx.y << 7, col0 = blockIdx.x << 7;
  v8f acc[4][2] = {};
  gemm_tile<false>(A, lda, row0, Bm, ldb, col0, kdim, sA0, sB0, sA1, sB1, acc);
  EPILOGUE_COORDS
  #pragma unroll
  for (int i = 0; i < 4; ++i)
    #pragma unroll
    for (int j = 0; j < 2; ++j)
      #pragma unroll
      for (int v = 0; v < 8; ++v)
        C[(size_t)(row0 + r0 + (i << 4) + v) * ldc + col0 + c0 + (j << 4)] =
            acc[i][j][v];
}

// ---------------------------------------------------------------------------
// Kernel: scores = Q @ K^T * scale + causal mask  (raw, pre-softmax, f32)
// grid = (S/128, S/128, B*NH). Tiles fully above the diagonal are skipped.
// ---------------------------------------------------------------------------
__global__ void __launch_bounds__(256)
scores_kernel(const bf16* __restrict__ qb, const bf16* __restrict__ kb,
              float* __restrict__ attn) {
  int bh = blockIdx.z;
  int q0 = blockIdx.y << 7, k0c = blockIdx.x << 7;
  if (k0c > q0 + 127) return;          // fully masked tile (uniform exit)
  GEMM_SMEM
  const bf16* Ap = qb + (size_t)bh * kS * kHD;
  const bf16* Bp = kb + (size_t)((bh >> 4) * kNKV + ((bh & 15) >> 2)) * kS * kHD;
  v8f acc[4][2] = {};
  gemm_tile<false>(Ap, kHD, q0, Bp, kHD, k0c, kHD, sA0, sB0, sA1, sB1, acc);
  float* Cp = attn + (size_t)bh * kS * kS;
  EPILOGUE_COORDS
  #pragma unroll
  for (int i = 0; i < 4; ++i)
    #pragma unroll
    for (int j = 0; j < 2; ++j)
      #pragma unroll
      for (int v = 0; v < 8; ++v) {
        int row = q0 + r0 + (i << 4) + v;
        int col = k0c + c0 + (j << 4);
        float val = acc[i][j][v] * SCALE + ((col > row) ? -1.0e9f : 0.0f);
        Cp[(size_t)row * kS + col] = val;
      }
}

// ---------------------------------------------------------------------------
// Kernel: row softmax in-place over attn (length S rows); causal rows only
// need k <= q; k > q is written as exact 0. grid = B*NH*S blocks of 256.
// ---------------------------------------------------------------------------
__global__ void __launch_bounds__(256)
softmax_kernel(float* __restrict__ attn) {
  __shared__ float red[256];
  size_t row = blockIdx.x;
  int q = (int)(row & (kS - 1));
  float* p = attn + row * (size_t)kS;
  int tid = threadIdx.x;
  int n = q + 1;
  float m = -INFINITY;
  for (int k = tid; k < n; k += 256) m = fmaxf(m, p[k]);
  red[tid] = m; __syncthreads();
  for (int s2 = 128; s2 > 0; s2 >>= 1) {
    if (tid < s2) red[tid] = fmaxf(red[tid], red[tid + s2]);
    __syncthreads();
  }
  m = red[0]; __syncthreads();
  float sum = 0.0f;
  for (int k = tid; k < n; k += 256) sum += __expf(p[k] - m);
  red[tid] = sum; __syncthreads();
  for (int s2 = 128; s2 > 0; s2 >>= 1) {
    if (tid < s2) red[tid] += red[tid + s2];
    __syncthreads();
  }
  float inv = 1.0f / red[0];
  for (int k = tid; k < n; k += 256) p[k] = __expf(p[k] - m) * inv;
  for (int k = n + tid; k < kS; k += 256) p[k] = 0.0f;
}

// ---------------------------------------------------------------------------
// Kernel: out_head = attn @ V. A = attn (f32, converted while staging),
// B = V^T stored [d, s] so the contraction dim is contiguous.
// Causality bounds the K loop at q0+128. grid = (1, S/128, B*NH).
// ---------------------------------------------------------------------------
__global__ void __launch_bounds__(256)
pv_kernel(const float* __restrict__ attn, const bf16* __restrict__ vb,
          bf16* __restrict__ ob) {
  int bh = blockIdx.z;
  int q0 = blockIdx.y << 7;
  int d0 = blockIdx.x << 7;   // N = 128 -> always 0
  GEMM_SMEM
  const float* Ap = attn + (size_t)bh * kS * kS;
  const bf16* Bp = vb + (size_t)((bh >> 4) * kNKV + ((bh & 15) >> 2)) * kHD * kS;
  v8f acc[4][2] = {};
  int kdim = q0 + 128;        // attn[q,k]==0 for k>q
  gemm_tile<true>(Ap, kS, q0, Bp, kS, d0, kdim, sA0, sB0, sA1, sB1, acc);
  int b = bh >> 4, h = bh & 15;
  EPILOGUE_COORDS
  #pragma unroll
  for (int i = 0; i < 4; ++i)
    #pragma unroll
    for (int j = 0; j < 2; ++j)
      #pragma unroll
      for (int v = 0; v < 8; ++v) {
        int row = q0 + r0 + (i << 4) + v;
        int col = d0 + c0 + (j << 4);
        ob[(size_t)(b * kS + row) * (kNH * kHD) + h * kHD + col] =
            (bf16)acc[i][j][v];
      }
}

// ---------------------------------------------------------------------------
// Kernel: f32 -> bf16 conversion, 4 elements per thread.
// ---------------------------------------------------------------------------
__global__ void __launch_bounds__(256)
cvt4_kernel(const float* __restrict__ src, bf16* __restrict__ dst, int n4) {
  int i = blockIdx.x * 256 + threadIdx.x;
  if (i < n4) {
    v4f v = *(const v4f*)(src + (size_t)i * 4);
    v4bf o;
    o[0] = (bf16)v[0]; o[1] = (bf16)v[1]; o[2] = (bf16)v[2]; o[3] = (bf16)v[3];
    *(v4bf*)(dst + (size_t)i * 4) = o;
  }
}

// ---------------------------------------------------------------------------
// Kernel: RoPE + head split/transpose.
// qkv: [B*S, 3072] bf16 (Q | K | V). Writes qb [B,NH,S,HD], kb [B,NKV,S,HD],
// vb [B,NKV,HD,S] (V transposed for the PV GEMM).
// ---------------------------------------------------------------------------
__global__ void __launch_bounds__(256)
rope_kernel(const bf16* __restrict__ qkv,
            const float* __restrict__ cosp, const float* __restrict__ sinp,
            bf16* __restrict__ qb, bf16* __restrict__ kb, bf16* __restrict__ vb) {
  size_t idx = (size_t)blockIdx.x * 256 + threadIdx.x;  // B*S*3072 total
  int col = (int)(idx % 3072);
  size_t bs = idx / 3072;
  int s = (int)(bs & (kS - 1));
  int b = (int)(bs >> 11);
  const bf16* row = qkv + bs * 3072;
  if (col < 2048) {                       // Q + RoPE
    int h = col >> 7, d = col & 127;
    float t  = (float)row[col];
    float c  = cosp[((size_t)b * kS + s) * kHD + d];
    float sn = sinp[((size_t)b * kS + s) * kHD + d];
    float rot = (d < 64) ? -(float)row[col + 64] : (float)row[col - 64];
    qb[((size_t)(b * kNH + h) * kS + s) * kHD + d] = (bf16)(t * c + rot * sn);
  } else if (col < 2560) {                // K + RoPE
    int h = (col - 2048) >> 7, d = (col - 2048) & 127;
    float t  = (float)row[col];
    float c  = cosp[((size_t)b * kS + s) * kHD + d];
    float sn = sinp[((size_t)b * kS + s) * kHD + d];
    float rot = (d < 64) ? -(float)row[col + 64] : (float)row[col - 64];
    kb[((size_t)(b * kNKV + h) * kS + s) * kHD + d] = (bf16)(t * c + rot * sn);
  } else {                                // V -> transposed [d, s]
    int h = (col - 2560) >> 7, d = (col - 2560) & 127;
    vb[((size_t)(b * kNKV + h) * kHD + d) * kS + s] = (bf16)(float)row[col];
  }
}

// ---------------------------------------------------------------------------
extern "C" void kernel_launch(void* const* d_in, const int* in_sizes, int n_in,
                              void* d_out, int out_size, void* d_ws, size_t ws_size,
                              hipStream_t stream) {
  (void)in_sizes; (void)n_in; (void)out_size; (void)ws_size;
  const float* x    = (const float*)d_in[0];
  const float* cosp = (const float*)d_in[1];
  const float* sinp = (const float*)d_in[2];
  // d_in[3] = attention_mask (causal mask computed analytically instead)
  const float* wq   = (const float*)d_in[4];
  const float* wk   = (const float*)d_in[5];
  const float* wv   = (const float*)d_in[6];
  const float* wo   = (const float*)d_in[7];

  float* out  = (float*)d_out;                          // [B,S,2048]
  float* attn = out + (size_t)kB * kS * kNH * kHD;      // [B,NH,S,S]

  // Workspace carve-up (bf16 elements); total ~88 MB.
  bf16* ws = (bf16*)d_ws;
  size_t off = 0;
  bf16* xb   = ws + off; off += (size_t)kB * kS * kH;            // 8388608
  bf16* wqb  = ws + off; off += (size_t)kNH * kHD * kH;          // 4194304
  bf16* wkb  = ws + off; off += (size_t)kNKV * kHD * kH;         // 1048576
  bf16* wvb  = ws + off; off += (size_t)kNKV * kHD * kH;         // 1048576
  bf16* wob  = ws + off; off += (size_t)kH * kNH * kHD;          // 4194304
  bf16* qb   = ws + off; off += (size_t)kB * kNH * kS * kHD;     // 8388608
  bf16* kb   = ws + off; off += (size_t)kB * kNKV * kS * kHD;    // 2097152
  bf16* vb   = ws + off; off += (size_t)kB * kNKV * kHD * kS;    // 2097152
  bf16* qkv  = ws + off; off += (size_t)kB * kS * 3072;          // 12582912
  bf16* ob   = qkv;   // reused after RoPE consumes qkv

  // 1) fp32 -> bf16 conversions
  cvt4_kernel<<<8192, 256, 0, stream>>>(x,  xb,  2097152);
  cvt4_kernel<<<4096, 256, 0, stream>>>(wq, wqb, 1048576);
  cvt4_kernel<<<1024, 256, 0, stream>>>(wk, wkb,  262144);
  cvt4_kernel<<<1024, 256, 0, stream>>>(wv, wvb,  262144);
  cvt4_kernel<<<4096, 256, 0, stream>>>(wo, wob, 1048576);

  // 2) QKV projections: [4096,2048] @ W^T -> qkv [4096, 3072]
  gemm_to_bf16<<<dim3(16, 32), 256, 0, stream>>>(xb, kH, wqb, kH, qkv,        3072, kH);
  gemm_to_bf16<<<dim3( 4, 32), 256, 0, stream>>>(xb, kH, wkb, kH, qkv + 2048, 3072, kH);
  gemm_to_bf16<<<dim3( 4, 32), 256, 0, stream>>>(xb, kH, wvb, kH, qkv + 2560, 3072, kH);

  // 3) RoPE + head transpose (V transposed to [d,s])
  rope_kernel<<<49152, 256, 0, stream>>>(qkv, cosp, sinp, qb, kb, vb);

  // 4) scores = Q@K^T * scale + causal mask (raw, into attn output region)
  scores_kernel<<<dim3(16, 16, kB * kNH), 256, 0, stream>>>(qb, kb, attn);

  // 5) softmax rows in place (zeros above diagonal)
  softmax_kernel<<<kB * kNH * kS, 256, 0, stream>>>(attn);

  // 6) out_head = attn @ V  -> ob [B*S, 2048] bf16
  pv_kernel<<<dim3(1, 16, kB * kNH), 256, 0, stream>>>(attn, vb, ob);

  // 7) out = ob @ wo^T -> f32
  gemm_to_f32<<<dim3(16, 32), 256, 0, stream>>>(ob, kNH * kHD, wob, kH, out, kH, kNH * kHD);
}
